// RWKV_Block_77945066488406
// MI455X (gfx1250) — matrix operand order repaired and checked by reading, not compiled, not run on hardware
//
#include <hip/hip_runtime.h>

typedef __bf16 bf16;
typedef __attribute__((ext_vector_type(8)))  bf16  v8bf;
typedef __attribute__((ext_vector_type(16))) bf16  v16bf;
typedef __attribute__((ext_vector_type(8)))  float v8f;
typedef __attribute__((ext_vector_type(4)))  int   v4i;
typedef __attribute__((address_space(1))) v4i gv4i;   // global b128 chunk
typedef __attribute__((address_space(3))) v4i lv4i;   // LDS b128 chunk

#define C_DIM 2048
#define L_DIM 512
#define B_DIM 8
#define H_DIM 32
#define S_DIM 64
#define NTOK  (B_DIM * L_DIM)
#define EPSV  1e-5f

#if defined(__has_builtin)
#if __has_builtin(__builtin_amdgcn_global_load_async_to_lds_b128) && \
    __has_builtin(__builtin_amdgcn_s_wait_asynccnt)
#define USE_ASYNC 1
#endif
#endif

// ---------------------------------------------------------------------------
// fp32 -> bf16 conversion (weights, each call; deterministic)
// ---------------------------------------------------------------------------
__global__ __launch_bounds__(256)
void cvt_bf16(const float* __restrict__ src, bf16* __restrict__ dst, int n) {
    int i = blockIdx.x * 256 + threadIdx.x;
    if (i < n) dst[i] = (bf16)src[i];
}

// ---------------------------------------------------------------------------
// WMMA bf16 GEMM:  Y[M,N] = epilogue( X[M,K] @ W[N,K]^T )
// block = 256 thr (8 waves); block tile 256(M) x 64(N); wave tile 32 x 64.
// B tile (64 rows x 64 K, bf16) staged in LDS, double-buffered with
// GLOBAL_LOAD_ASYNC_TO_LDS_B128 (ASYNCcnt) when available.
// A fragment (16-bit A 16x32): lane = row + 16*half,
//   elems 0..15 = K[half*16 .. +15] (matches B)
// B fragment (16-bit B 32x16): lane holds column lane&15,
//   elems 0..15 = K[(lane>>4)*16 .. +15] -> contiguous W row slice.
// D: vgpr e -> M = e + 8*(lane>>4), N = lane&15.
// act: 0 none, 1 silu, 2 sigmoid, 3 relu^2
// ---------------------------------------------------------------------------
#define KC   64            // K chunk staged per double-buffer step
#define BROW 80            // LDS row stride (elements): 160B, 32B-aligned, bank-spread

__global__ __launch_bounds__(256)
void gemm_bf16(const bf16* __restrict__ X, const bf16* __restrict__ W,
               float* __restrict__ outF, bf16* __restrict__ outB,
               const float* __restrict__ scale, const float* __restrict__ resid,
               int N, int K, int act)
{
    __shared__ bf16 sB[2][64 * BROW];      // 2 x 10 KB

    const int tid  = threadIdx.x;
    const int lane = tid & 31;
    const int wave = tid >> 5;
    const int mtile = blockIdx.x * 256 + wave * 32;
    const int nbase = blockIdx.y * 64;
    const int row = lane & 15;
    const int kh  = lane >> 4;

    const bf16* xrow0 = X + (size_t)(mtile + row) * K;
    const bf16* xrow1 = X + (size_t)(mtile + 16 + row) * K;

    // cooperative fill of one B buffer: 64 rows x 128B = 512 x 16B ops,
    // 2 ops per thread
    auto fill = [&](int buf, int k0) {
#pragma unroll
        for (int hh = 0; hh < 2; ++hh) {
            int o = tid + hh * 256;
            int r = o >> 3, c = o & 7;
            const bf16* gp = W + (size_t)(nbase + r) * K + k0 + c * 8;
            bf16* lp = &sB[buf][r * BROW + c * 8];
#ifdef USE_ASYNC
            __builtin_amdgcn_global_load_async_to_lds_b128(
                (gv4i*)(void*)gp, (lv4i*)lp, 0, 0);
#else
            *(v8bf*)lp = *(const v8bf*)gp;
#endif
        }
    };

    v8f acc0[4], acc1[4];
#pragma unroll
    for (int j = 0; j < 4; ++j)
#pragma unroll
        for (int e = 0; e < 8; ++e) { acc0[j][e] = 0.0f; acc1[j][e] = 0.0f; }

    int buf = 0;
    fill(0, 0);
#ifdef USE_ASYNC
    __builtin_amdgcn_s_wait_asynccnt(0);
#endif
    __syncthreads();

    for (int kc = 0; kc < K; kc += KC) {
        if (kc + KC < K) fill(buf ^ 1, kc + KC);     // prefetch next chunk

#pragma unroll
        for (int s = 0; s < 2; ++s) {                // two 32-K steps per chunk
            int kg = kc + s * 32;
            v16bf a0 = *(const v16bf*)(xrow0 + kg + kh * 16);
            v16bf a1 = *(const v16bf*)(xrow1 + kg + kh * 16);
#pragma unroll
            for (int j = 0; j < 4; ++j) {
                const bf16* bp = &sB[buf][(j * 16 + row) * BROW + s * 32 + kh * 16];
                v16bf b = *(const v16bf*)bp;
                acc0[j] = __builtin_amdgcn_wmma_f32_16x16x32_bf16(
                              false, a0, false, b, (short)0, acc0[j], false, false);
                acc1[j] = __builtin_amdgcn_wmma_f32_16x16x32_bf16(
                              false, a1, false, b, (short)0, acc1[j], false, false);
            }
        }
#ifdef USE_ASYNC
        __builtin_amdgcn_s_wait_asynccnt(0);
#endif
        __syncthreads();
        buf ^= 1;
    }

#pragma unroll
    for (int j = 0; j < 4; ++j) {
        int col = nbase + j * 16 + row;
#pragma unroll
        for (int e = 0; e < 8; ++e) {
#pragma unroll
            for (int half = 0; half < 2; ++half) {
                int m = mtile + half * 16 + e + kh * 8;
                size_t idx = (size_t)m * N + col;
                float val = half ? acc1[j][e] : acc0[j][e];
                if (act == 1)      val = val / (1.0f + __expf(-val));    // silu
                else if (act == 2) val = 1.0f / (1.0f + __expf(-val));   // sigmoid
                else if (act == 3) { val = fmaxf(val, 0.0f); val = val * val; }
                if (scale) val *= scale[idx];
                if (resid) val += resid[idx];
                if (outF) outF[idx] = val;
                if (outB) outB[idx] = (bf16)val;
            }
        }
    }
}

// ---------------------------------------------------------------------------
// Time-mix prep: LN1 + token shift + 160-wide LoRA + decay LoRA.
// One block per token (256 threads).
// ---------------------------------------------------------------------------
__global__ __launch_bounds__(256)
void tmix_prep(const float* __restrict__ x, const float* __restrict__ state,
               const int* __restrict__ ip,
               const float* __restrict__ ln1_w, const float* __restrict__ ln1_b,
               const float* __restrict__ maa_x, const float* __restrict__ maa_w,
               const float* __restrict__ maa_k, const float* __restrict__ maa_v,
               const float* __restrict__ maa_r, const float* __restrict__ maa_g,
               const float* __restrict__ maa_w1, const float* __restrict__ maa_w2,
               const float* __restrict__ time_decay,
               const float* __restrict__ decay_w1, const float* __restrict__ decay_w2,
               bf16* __restrict__ xk_o, bf16* __restrict__ xv_o,
               bf16* __restrict__ xr_o, bf16* __restrict__ xg_o,
               float* __restrict__ w_o)
{
    const int C = C_DIM;
    int tok = blockIdx.x, tid = threadIdx.x;
    int b = tok / L_DIM, l = tok % L_DIM;
    int i0 = (2 + S_DIM) * ip[0];

    __shared__ float s_xa[C_DIM], s_sx[C_DIM], s_tmp[C_DIM];
    __shared__ float s_t[160], s_dw[64];
    __shared__ float red[256], red2[256];

    const float* xc = x + (size_t)tok * C;

    // ---- LN of current token ----
    float sum = 0.f, sq = 0.f;
    for (int c = tid; c < C; c += 256) { float v = xc[c]; sum += v; sq += v * v; }
    red[tid] = sum; red2[tid] = sq; __syncthreads();
    for (int o = 128; o > 0; o >>= 1) {
        if (tid < o) { red[tid] += red[tid + o]; red2[tid] += red2[tid + o]; }
        __syncthreads();
    }
    float m = red[0] / C;
    float rstd = rsqrtf(red2[0] / C - m * m + EPSV);
    __syncthreads();
    for (int c = tid; c < C; c += 256)
        s_xa[c] = (xc[c] - m) * rstd * ln1_w[c] + ln1_b[c];

    // ---- token shift (prev token LN, or raw state row at l==0) ----
    if (l > 0) {
        const float* xp = xc - C;
        sum = 0.f; sq = 0.f;
        for (int c = tid; c < C; c += 256) { float v = xp[c]; sum += v; sq += v * v; }
        red[tid] = sum; red2[tid] = sq; __syncthreads();
        for (int o = 128; o > 0; o >>= 1) {
            if (tid < o) { red[tid] += red[tid + o]; red2[tid] += red2[tid + o]; }
            __syncthreads();
        }
        float m2 = red[0] / C;
        float rstd2 = rsqrtf(red2[0] / C - m2 * m2 + EPSV);
        __syncthreads();
        for (int c = tid; c < C; c += 256)
            s_sx[c] = ((xp[c] - m2) * rstd2 * ln1_w[c] + ln1_b[c]) - s_xa[c];
    } else {
        const float* sp = state + ((size_t)b * (2 + S_DIM) + i0 + 1) * C;
        for (int c = tid; c < C; c += 256) s_sx[c] = sp[c] - s_xa[c];
    }
    __syncthreads();

    // ---- xxx = xa + sx*maa_x  (staged in s_tmp) ----
    for (int c = tid; c < C; c += 256) s_tmp[c] = s_xa[c] + s_sx[c] * maa_x[c];
    __syncthreads();

    // ---- t = tanh(xxx @ maa_w1)   (160 outputs) ----
    if (tid < 160) {
        float a = 0.f;
        for (int c = 0; c < C; ++c) a += s_tmp[c] * maa_w1[c * 160 + tid];
        s_t[tid] = tanhf(a);
    }
    __syncthreads();

    // ---- per-channel mixes; xw kept in s_tmp for the decay LoRA ----
    for (int c = tid; c < C; c += 256) {
        float xa = s_xa[c], sx = s_sx[c];
        float mf[5];
#pragma unroll
        for (int f = 0; f < 5; ++f) {
            float a = 0.f;
#pragma unroll
            for (int d = 0; d < 32; ++d)
                a += s_t[f * 32 + d] * maa_w2[(size_t)(f * 32 + d) * C + c];
            mf[f] = a;
        }
        size_t o = (size_t)tok * C + c;
        float xw = xa + sx * (maa_w[c] + mf[0]);
        xk_o[o] = (bf16)(xa + sx * (maa_k[c] + mf[1]));
        xv_o[o] = (bf16)(xa + sx * (maa_v[c] + mf[2]));
        xr_o[o] = (bf16)(xa + sx * (maa_r[c] + mf[3]));
        xg_o[o] = (bf16)(xa + sx * (maa_g[c] + mf[4]));
        s_tmp[c] = xw;
    }
    __syncthreads();

    // ---- decay: w = exp(-exp(time_decay + tanh(xw@dw1)@dw2)) ----
    if (tid < 64) {
        float a = 0.f;
        for (int c = 0; c < C; ++c) a += s_tmp[c] * decay_w1[c * 64 + tid];
        s_dw[tid] = tanhf(a);
    }
    __syncthreads();
    for (int c = tid; c < C; c += 256) {
        float a = time_decay[c];
#pragma unroll
        for (int d = 0; d < 64; ++d) a += s_dw[d] * decay_w2[(size_t)d * C + c];
        w_o[(size_t)tok * C + c] = __expf(-__expf(a));
    }
}

// ---------------------------------------------------------------------------
// WKV scan: block per (b,h), 64 threads; thread t owns state column t.
// ---------------------------------------------------------------------------
__global__ __launch_bounds__(64)
void wkv_scan(const float* __restrict__ r, const float* __restrict__ k,
              const float* __restrict__ v, const float* __restrict__ w,
              const float* __restrict__ faaaa, const float* __restrict__ state,
              const int* __restrict__ ip, float* __restrict__ y)
{
    const int C = C_DIM;
    int bh = blockIdx.x;
    int b = bh / H_DIM, h = bh % H_DIM;
    int t = threadIdx.x;
    int i0 = (2 + S_DIM) * ip[0];

    float st[64];
    const float* sb = state + ((size_t)b * (2 + S_DIM) + i0 + 2) * C;
#pragma unroll
    for (int s = 0; s < 64; ++s)
        st[s] = sb[(size_t)(2 * h + (s >> 5)) * C + (s & 31) * 64 + t];

    __shared__ float sk[64], sw[64], sr[64], sf[64];
    sf[t] = faaaa[h * 64 + t];
    __syncthreads();

    for (int l = 0; l < L_DIM; ++l) {
        size_t base = ((size_t)(b * L_DIM + l)) * C + h * 64;
        float kv = k[base + t], wv = w[base + t];
        float rv = r[base + t], vv = v[base + t];
        sk[t] = kv; sw[t] = wv; sr[t] = rv;
        __syncthreads();
        float fsum = 0.f;
#pragma unroll
        for (int s = 0; s < 64; ++s) fsum += sr[s] * sf[s] * sk[s];
        float out = vv * fsum;
#pragma unroll
        for (int s = 0; s < 64; ++s) {
            out += sr[s] * st[s];
            st[s] = sk[s] * vv + sw[s] * st[s];
        }
        y[base + t] = out;
        __syncthreads();
    }
}

// ---------------------------------------------------------------------------
// Group norm (per head) * silu-gate -> bf16 for Wo GEMM
// ---------------------------------------------------------------------------
__global__ __launch_bounds__(64)
void gnorm_gate(const float* __restrict__ y, const float* __restrict__ g,
                const float* __restrict__ gn_w, const float* __restrict__ gn_b,
                bf16* __restrict__ z)
{
    int blh = blockIdx.x;
    int h = blh % H_DIM, tok = blh / H_DIM;
    int t = threadIdx.x;
    __shared__ float red[64];

    size_t base = (size_t)tok * C_DIM + h * 64;
    float val = y[base + t];

    red[t] = val; __syncthreads();
    for (int o = 32; o > 0; o >>= 1) { if (t < o) red[t] += red[t + o]; __syncthreads(); }
    float m = red[0] / 64.f; __syncthreads();
    float d = val - m;
    red[t] = d * d; __syncthreads();
    for (int o = 32; o > 0; o >>= 1) { if (t < o) red[t] += red[t + o]; __syncthreads(); }
    float var = red[0] / 64.f;
    float nv = d * rsqrtf(var + EPSV);
    int c = h * 64 + t;
    z[base + t] = (bf16)((nv * gn_w[c] + gn_b[c]) * g[base + t]);
}

// ---------------------------------------------------------------------------
// Channel-mix prep: LN2 + token shift -> bf16 xk2 / xr2
// ---------------------------------------------------------------------------
__global__ __launch_bounds__(256)
void cmix_prep(const float* __restrict__ x2, const float* __restrict__ state,
               const int* __restrict__ ip,
               const float* __restrict__ ln2_w, const float* __restrict__ ln2_b,
               const float* __restrict__ fmaa_k, const float* __restrict__ fmaa_r,
               bf16* __restrict__ xk2, bf16* __restrict__ xr2)
{
    const int C = C_DIM;
    int tok = blockIdx.x, tid = threadIdx.x;
    int b = tok / L_DIM, l = tok % L_DIM;
    int i0 = (2 + S_DIM) * ip[0];

    __shared__ float s_xc[C_DIM];
    __shared__ float red[256], red2[256];

    const float* xc = x2 + (size_t)tok * C;

    float sum = 0.f, sq = 0.f;
    for (int c = tid; c < C; c += 256) { float v = xc[c]; sum += v; sq += v * v; }
    red[tid] = sum; red2[tid] = sq; __syncthreads();
    for (int o = 128; o > 0; o >>= 1) {
        if (tid < o) { red[tid] += red[tid + o]; red2[tid] += red2[tid + o]; }
        __syncthreads();
    }
    float m = red[0] / C;
    float rstd = rsqrtf(red2[0] / C - m * m + EPSV);
    __syncthreads();
    for (int c = tid; c < C; c += 256)
        s_xc[c] = (xc[c] - m) * rstd * ln2_w[c] + ln2_b[c];

    float m2 = 0.f, rstd2 = 1.f;
    const float* xp = xc - C;
    if (l > 0) {
        sum = 0.f; sq = 0.f;
        for (int c = tid; c < C; c += 256) { float v = xp[c]; sum += v; sq += v * v; }
        red[tid] = sum; red2[tid] = sq; __syncthreads();
        for (int o = 128; o > 0; o >>= 1) {
            if (tid < o) { red[tid] += red[tid + o]; red2[tid] += red2[tid + o]; }
            __syncthreads();
        }
        m2 = red[0] / C;
        rstd2 = rsqrtf(red2[0] / C - m2 * m2 + EPSV);
    }
    __syncthreads();

    const float* sp = state + ((size_t)b * (2 + S_DIM) + i0) * C;
    for (int c = tid; c < C; c += 256) {
        float pv = (l > 0) ? ((xp[c] - m2) * rstd2 * ln2_w[c] + ln2_b[c]) : sp[c];
        float sxc = pv - s_xc[c];
        size_t o = (size_t)tok * C + c;
        xk2[o] = (bf16)(s_xc[c] + sxc * fmaa_k[c]);
        xr2[o] = (bf16)(s_xc[c] + sxc * fmaa_r[c]);
    }
}

// ---------------------------------------------------------------------------
extern "C" void kernel_launch(void* const* d_in, const int* in_sizes, int n_in,
                              void* d_out, int out_size, void* d_ws, size_t ws_size,
                              hipStream_t stream)
{
    (void)in_sizes; (void)n_in; (void)out_size; (void)ws_size;

    const float* x        = (const float*)d_in[0];
    const float* state    = (const float*)d_in[1];
    const int*   ip       = (const int*)d_in[2];
    const float* ln1_w    = (const float*)d_in[3];
    const float* ln1_b    = (const float*)d_in[4];
    const float* ln2_w    = (const float*)d_in[5];
    const float* ln2_b    = (const float*)d_in[6];
    const float* maa_x    = (const float*)d_in[7];
    const float* maa_w    = (const float*)d_in[8];
    const float* maa_k    = (const float*)d_in[9];
    const float* maa_v    = (const float*)d_in[10];
    const float* maa_r    = (const float*)d_in[11];
    const float* maa_g    = (const float*)d_in[12];
    const float* maa_w1   = (const float*)d_in[13];
    const float* maa_w2   = (const float*)d_in[14];
    const float* time_dec = (const float*)d_in[15];
    const float* decay_w1 = (const float*)d_in[16];
    const float* decay_w2 = (const float*)d_in[17];
    const float* faaaa    = (const float*)d_in[18];
    const float* Wr       = (const float*)d_in[19];
    const float* Wk       = (const float*)d_in[20];
    const float* Wv       = (const float*)d_in[21];
    const float* Wg       = (const float*)d_in[22];
    const float* Wo       = (const float*)d_in[23];
    const float* gn_w     = (const float*)d_in[24];
    const float* gn_b     = (const float*)d_in[25];
    const float* fmaa_k   = (const float*)d_in[26];
    const float* fmaa_r   = (const float*)d_in[27];
    const float* fWk      = (const float*)d_in[28];
    const float* fWr      = (const float*)d_in[29];
    const float* fWv      = (const float*)d_in[30];

    // ---- workspace layout (with deterministic aliasing) ----
    char* p = (char*)d_ws;
    const size_t WB   = (size_t)C_DIM * C_DIM * 2;   // bf16 weight
    const size_t FB16 = (size_t)NTOK * C_DIM * 2;    // bf16 activation
    const size_t FB32 = (size_t)NTOK * C_DIM * 4;    // fp32 activation
    bf16* Wbf[8];
    for (int i = 0; i < 8; ++i) Wbf[i] = (bf16*)(p + (size_t)i * WB);
    size_t off = 8 * WB;
    bf16*  xk_bf = (bf16*)(p + off); off += FB16;
    bf16*  xv_bf = (bf16*)(p + off); off += FB16;
    bf16*  xr_bf = (bf16*)(p + off); off += FB16;
    bf16*  xg_bf = (bf16*)(p + off); off += FB16;
    float* w_f   = (float*)(p + off); off += FB32;
    float* r_f   = (float*)(p + off); off += FB32;
    float* k_f   = (float*)(p + off); off += FB32;
    float* v_f   = (float*)(p + off); off += FB32;
    float* g_f   = (float*)(p + off); off += FB32;
    float* y_f   = (float*)(p + off); off += FB32;
    // aliases over dead regions:
    bf16*  z_bf   = xk_bf;   // xk dead after k GEMM
    float* x2_f   = r_f;     // r dead after scan
    bf16*  xk2_bf = xv_bf;   // xv dead after v GEMM
    bf16*  xr2_bf = xr_bf;   // xr dead after r GEMM
    float* rr_f   = k_f;     // k dead after scan
    bf16*  kk_bf  = xg_bf;   // xg dead after g GEMM

    // ---- 1. convert weights to bf16 ----
    const int nW = C_DIM * C_DIM;
    const float* wsrc[8] = { Wr, Wk, Wv, Wg, Wo, fWr, fWk, fWv };
    for (int i = 0; i < 8; ++i)
        cvt_bf16<<<nW / 256, 256, 0, stream>>>(wsrc[i], Wbf[i], nW);

    // ---- 2. time-mix prep ----
    tmix_prep<<<NTOK, 256, 0, stream>>>(x, state, ip, ln1_w, ln1_b,
        maa_x, maa_w, maa_k, maa_v, maa_r, maa_g, maa_w1, maa_w2,
        time_dec, decay_w1, decay_w2, xk_bf, xv_bf, xr_bf, xg_bf, w_f);

    // ---- 3. r/k/v/g GEMMs (WMMA bf16, LDS double-buffered B) ----
    dim3 gg(NTOK / 256, C_DIM / 64);
    gemm_bf16<<<gg, 256, 0, stream>>>(xr_bf, Wbf[0], r_f, nullptr, nullptr, nullptr, C_DIM, C_DIM, 0);
    gemm_bf16<<<gg, 256, 0, stream>>>(xk_bf, Wbf[1], k_f, nullptr, nullptr, nullptr, C_DIM, C_DIM, 0);
    gemm_bf16<<<gg, 256, 0, stream>>>(xv_bf, Wbf[2], v_f, nullptr, nullptr, nullptr, C_DIM, C_DIM, 0);
    gemm_bf16<<<gg, 256, 0, stream>>>(xg_bf, Wbf[3], g_f, nullptr, nullptr, nullptr, C_DIM, C_DIM, 1);

    // ---- 4. WKV scan ----
    wkv_scan<<<B_DIM * H_DIM, 64, 0, stream>>>(r_f, k_f, v_f, w_f, faaaa, state, ip, y_f);

    // ---- 5. group norm * gate -> bf16 ----
    gnorm_gate<<<NTOK * H_DIM, 64, 0, stream>>>(y_f, g_f, gn_w, gn_b, z_bf);

    // ---- 6. output projection with residual: x2 = x + z @ Wo^T ----
    gemm_bf16<<<gg, 256, 0, stream>>>(z_bf, Wbf[4], x2_f, nullptr, nullptr, x, C_DIM, C_DIM, 0);

    // ---- 7. channel-mix prep ----
    cmix_prep<<<NTOK, 256, 0, stream>>>(x2_f, state, ip, ln2_w, ln2_b,
                                        fmaa_k, fmaa_r, xk2_bf, xr2_bf);

    // ---- 8-10. channel-mix GEMMs ----
    gemm_bf16<<<gg, 256, 0, stream>>>(xr2_bf, Wbf[5], rr_f, nullptr, nullptr, nullptr, C_DIM, C_DIM, 2);
    gemm_bf16<<<gg, 256, 0, stream>>>(xk2_bf, Wbf[6], nullptr, kk_bf, nullptr, nullptr, C_DIM, C_DIM, 3);
    gemm_bf16<<<gg, 256, 0, stream>>>(kk_bf, Wbf[7], (float*)d_out, nullptr, rr_f, x2_f, C_DIM, C_DIM, 0);
}
